// SAModule_34024730919176
// MI455X (gfx1250) — compile-verified, matrix operand order checked by reading, not run on hardware
//
#include <hip/hip_runtime.h>

typedef __attribute__((ext_vector_type(16))) _Float16 v16h;
typedef __attribute__((ext_vector_type(8)))  float    v8f;

#define NPTS    8192
#define BATCHSZ 1024
#define KNN     64
#define RAD2    0.04f
#define BN_EPS  1e-5f
#define LSTRIDE 80   // halfs per row: 64 data + 16 pad => 160B rows, 32B aligned
#define PPB     8    // points per block (amortizes weight/pos staging)

__launch_bounds__(128)
__global__ void sa_module_fused_kernel(
    const float* __restrict__ pos,
    const float* __restrict__ W1, const float* __restrict__ b1,
    const float* __restrict__ g1, const float* __restrict__ be1,
    const float* __restrict__ m1, const float* __restrict__ v1,
    const float* __restrict__ W2, const float* __restrict__ b2,
    const float* __restrict__ g2, const float* __restrict__ be2,
    const float* __restrict__ m2, const float* __restrict__ v2,
    const float* __restrict__ W3, const float* __restrict__ b3,
    const float* __restrict__ g3, const float* __restrict__ be3,
    const float* __restrict__ m3, const float* __restrict__ v3,
    float* __restrict__ out)
{
    __shared__ float spos[BATCHSZ * 3];                  // batch positions
    alignas(32) __shared__ _Float16 sWt2[64  * LSTRIDE]; // W2^T [n][k] f16
    alignas(32) __shared__ _Float16 sWt3[128 * LSTRIDE]; // W3^T [n][k] f16
    alignas(32) __shared__ _Float16 sH1 [64  * LSTRIDE]; // layer1 act f16
    alignas(32) __shared__ _Float16 sH2 [64  * LSTRIDE]; // layer2 act f16
    __shared__ float sW1[3 * 64];
    __shared__ float sB1[64],  sS1[64],  sT1[64];
    __shared__ float sB2[64],  sS2[64],  sT2[64];
    __shared__ float sB3[128], sS3[128], sT3[128];
    __shared__ float sRel[KNN][4];
    __shared__ int   sNbr[KNN];
    __shared__ int   sWsum[4];
    __shared__ float sMax[4 * 128];

    const int tid       = threadIdx.x;
    const int wave      = tid >> 5;
    const int lane      = tid & 31;
    const int firstPt   = blockIdx.x * PPB;                 // 8 pts, same cloud
    const int batchBase = (firstPt / BATCHSZ) * BATCHSZ;

    // ---- Stage 0 (once per block): weights + folded BN + batch positions ----
    for (int e = tid; e < 64; e += 128) {
        float s1v = g1[e] * rsqrtf(v1[e] + BN_EPS);
        sB1[e] = b1[e]; sS1[e] = s1v; sT1[e] = be1[e] - m1[e] * s1v;
        float s2v = g2[e] * rsqrtf(v2[e] + BN_EPS);
        sB2[e] = b2[e]; sS2[e] = s2v; sT2[e] = be2[e] - m2[e] * s2v;
    }
    for (int e = tid; e < 128; e += 128) {
        float s3v = g3[e] * rsqrtf(v3[e] + BN_EPS);
        sB3[e] = b3[e]; sS3[e] = s3v; sT3[e] = be3[e] - m3[e] * s3v;
    }
    for (int e = tid; e < 3 * 64; e += 128) sW1[e] = W1[e];
    for (int e = tid; e < 64 * 64; e += 128) {           // W2 [k][n] -> sWt2[n][k]
        int k = e >> 6, n = e & 63;
        sWt2[n * LSTRIDE + k] = (_Float16)W2[e];
    }
    for (int e = tid; e < 64 * 128; e += 128) {          // W3 [k][n] -> sWt3[n][k]
        int k = e >> 7, n = e & 127;
        sWt3[n * LSTRIDE + k] = (_Float16)W3[e];
    }
    for (int e = tid; e < BATCHSZ * 3; e += 128) spos[e] = pos[(size_t)batchBase * 3 + e];
    __syncthreads();

    const int rowBase = wave * 16;
    const int lm      = lane & 15;          // M (A) / N (B,D) within tile
    const int kb      = (lane >> 4) * 16;   // K-half selector (halfs)
    const int rowOff  = (lane >> 4) << 3;   // D rows: lanes>=16 hold M+8

    for (int p = 0; p < PPB; ++p) {
        const int   i  = firstPt + p;
        const int   li = i - batchBase;
        const float px = spos[li * 3 + 0];
        const float py = spos[li * 3 + 1];
        const float pz = spos[li * 3 + 2];

        // ---- Stage 1: deterministic ball query (first-K in index order) ----
        const int j0 = tid * (BATCHSZ / 128);   // 8 contiguous candidates/thread
        int myCnt = 0;
        for (int j = j0; j < j0 + (BATCHSZ / 128); ++j) {
            float dx = spos[j * 3 + 0] - px;
            float dy = spos[j * 3 + 1] - py;
            float dz = spos[j * 3 + 2] - pz;
            if (dx * dx + dy * dy + dz * dz <= RAD2) ++myCnt;
        }
        // wave32 inclusive scan (deterministic)
        int inc = myCnt;
#pragma unroll
        for (int d = 1; d < 32; d <<= 1) {
            int n = __shfl_up(inc, d, 32);
            if (lane >= d) inc += n;
        }
        if (lane == 31) sWsum[wave] = inc;      // wave total
        __syncthreads();
        int off = inc - myCnt;                  // exclusive within wave
        for (int w = 0; w < 4; ++w) if (w < wave) off += sWsum[w];
        const int total0 = sWsum[0] + sWsum[1] + sWsum[2] + sWsum[3];
        for (int j = j0; j < j0 + (BATCHSZ / 128); ++j) {
            float dx = spos[j * 3 + 0] - px;
            float dy = spos[j * 3 + 1] - py;
            float dz = spos[j * 3 + 2] - pz;
            if (dx * dx + dy * dy + dz * dz <= RAD2) {
                if (off < KNN) sNbr[off] = j;   // local index within batch
                ++off;
            }
        }
        __syncthreads();
        {
            int total = total0 < KNN ? total0 : KNN;
            for (int e = total + tid; e < KNN; e += 128) sNbr[e] = li; // pad self
        }
        __syncthreads();

        // ---- Stage 2: relative positions ----
        if (tid < KNN) {
            int j = sNbr[tid];
            sRel[tid][0] = spos[j * 3 + 0] - px;
            sRel[tid][1] = spos[j * 3 + 1] - py;
            sRel[tid][2] = spos[j * 3 + 2] - pz;
        }
        __syncthreads();

        // ---- Stage 3: layer 1 (3 -> 64), scalar f32 -> f16 ----
        for (int e = tid; e < 64 * 64; e += 128) {
            int kk = e >> 6, c = e & 63;
            float a = sB1[c]
                    + sRel[kk][0] * sW1[c]
                    + sRel[kk][1] * sW1[64 + c]
                    + sRel[kk][2] * sW1[128 + c];
            a = fmaxf(a, 0.0f);
            sH1[kk * LSTRIDE + c] = (_Float16)(sS1[c] * a + sT1[c]);
        }
        __syncthreads();

        // ---- Stage 4a: layer 2 WMMA [16x64]x[64x64], per-wave row strip ----
        {
            v16h a0 = *(const v16h*)&sH1[(rowBase + lm) * LSTRIDE + kb];
            v16h a1 = *(const v16h*)&sH1[(rowBase + lm) * LSTRIDE + kb + 32];
#pragma unroll
            for (int t = 0; t < 4; ++t) {
                int n = t * 16 + lm;
                v16h bv0 = *(const v16h*)&sWt2[n * LSTRIDE + kb];
                v16h bv1 = *(const v16h*)&sWt2[n * LSTRIDE + kb + 32];
                v8f acc = {};
                acc = __builtin_amdgcn_wmma_f32_16x16x32_f16(false, a0, false, bv0,
                                                             (short)0, acc, false, false);
                acc = __builtin_amdgcn_wmma_f32_16x16x32_f16(false, a1, false, bv1,
                                                             (short)0, acc, false, false);
                float bb = sB2[n], ss = sS2[n], tt = sT2[n];
#pragma unroll
                for (int v = 0; v < 8; ++v) {
                    int row = rowBase + v + rowOff;
                    float y = fmaxf(acc[v] + bb, 0.0f);
                    sH2[row * LSTRIDE + n] = (_Float16)(ss * y + tt);
                }
            }
        }
        // wave reads back only its own rows; per-wave LDS ordering suffices

        // ---- Stage 4b: layer 3 WMMA [16x64]x[64x128] + fused row-max ----
        {
            v16h a0 = *(const v16h*)&sH2[(rowBase + lm) * LSTRIDE + kb];
            v16h a1 = *(const v16h*)&sH2[(rowBase + lm) * LSTRIDE + kb + 32];
#pragma unroll
            for (int t = 0; t < 8; ++t) {
                int n = t * 16 + lm;
                v16h bv0 = *(const v16h*)&sWt3[n * LSTRIDE + kb];
                v16h bv1 = *(const v16h*)&sWt3[n * LSTRIDE + kb + 32];
                v8f acc = {};
                acc = __builtin_amdgcn_wmma_f32_16x16x32_f16(false, a0, false, bv0,
                                                             (short)0, acc, false, false);
                acc = __builtin_amdgcn_wmma_f32_16x16x32_f16(false, a1, false, bv1,
                                                             (short)0, acc, false, false);
                float bb = sB3[n], ss = sS3[n], tt = sT3[n];
                float pm = -3.402823466e38f;
#pragma unroll
                for (int v = 0; v < 8; ++v) {
                    float y = fmaxf(acc[v] + bb, 0.0f);
                    pm = fmaxf(pm, ss * y + tt);
                }
                pm = fmaxf(pm, __shfl_xor(pm, 16, 32)); // merge 8-row halves
                if (lane < 16) sMax[wave * 128 + n] = pm;
            }
        }
        __syncthreads();

        // ---- Stage 5: cross-wave max, write [128] f32 ----
        {
            float r = fmaxf(fmaxf(sMax[tid], sMax[128 + tid]),
                            fmaxf(sMax[256 + tid], sMax[384 + tid]));
            out[(size_t)i * 128 + tid] = r;
        }
        __syncthreads();   // protect sMax/sNbr/sRel/sH1 before next iteration
    }
}

extern "C" void kernel_launch(void* const* d_in, const int* in_sizes, int n_in,
                              void* d_out, int out_size, void* d_ws, size_t ws_size,
                              hipStream_t stream) {
    (void)in_sizes; (void)n_in; (void)out_size; (void)d_ws; (void)ws_size;
    // setup_inputs order: 0:x(unused) 1:pos 2:batch(unused, batch=i/1024)
    //                     3..8:  W1,b1,g1,be1,m1,v1
    //                     9..14: W2,b2,g2,be2,m2,v2
    //                     15..20:W3,b3,g3,be3,m3,v3
    const float* pos = (const float*)d_in[1];
    const float* W1  = (const float*)d_in[3];
    const float* b1  = (const float*)d_in[4];
    const float* g1  = (const float*)d_in[5];
    const float* be1 = (const float*)d_in[6];
    const float* m1  = (const float*)d_in[7];
    const float* v1  = (const float*)d_in[8];
    const float* W2  = (const float*)d_in[9];
    const float* b2  = (const float*)d_in[10];
    const float* g2  = (const float*)d_in[11];
    const float* be2 = (const float*)d_in[12];
    const float* m2  = (const float*)d_in[13];
    const float* v2  = (const float*)d_in[14];
    const float* W3  = (const float*)d_in[15];
    const float* b3  = (const float*)d_in[16];
    const float* g3  = (const float*)d_in[17];
    const float* be3 = (const float*)d_in[18];
    const float* m3  = (const float*)d_in[19];
    const float* v3  = (const float*)d_in[20];

    sa_module_fused_kernel<<<NPTS / PPB, 128, 0, stream>>>(
        pos,
        W1, b1, g1, be1, m1, v1,
        W2, b2, g2, be2, m2, v2,
        W3, b3, g3, be3, m3, v3,
        (float*)d_out);
}
